// FastFFN_32143535243830
// MI455X (gfx1250) — matrix operand order repaired and checked by reading, not compiled, not run on hardware
//
#include <hip/hip_runtime.h>
#include <hip/hip_bf16.h>

typedef __attribute__((ext_vector_type(16))) __bf16 v16bf;
typedef __attribute__((ext_vector_type(8)))  __bf16 v8bf;
typedef __attribute__((ext_vector_type(8)))  float  v8f;

#define DD   1024
#define HH   512
#define NLV  8
#define NTOK 8192   // B*T = 4*2048

__device__ __forceinline__ v16bf cat8(v8bf lo, v8bf hi) {
    return __builtin_shufflevector(lo, hi, 0,1,2,3,4,5,6,7,8,9,10,11,12,13,14,15);
}

// ---------------------------------------------------------------------------
// Weight prep: fp32 -> bf16, transposed K-contiguous so a WMMA B-fragment is
// two b128 loads per lane.  w1t[l][col(1024: a|b)][k(1024)], w2t[l][col][k(512)]
// ---------------------------------------------------------------------------
__global__ void prep_w1(const float* __restrict__ w1a, const float* __restrict__ w1b,
                        __bf16* __restrict__ w1t) {
    int idx = blockIdx.x * 256 + threadIdx.x;      // 8*1024*1024 total
    int l   = idx >> 20;
    int rem = idx & 0xFFFFF;
    int col = rem >> 10;
    int k   = rem & 1023;
    float v = (col < HH) ? w1a[((size_t)l * DD + k) * HH + col]
                         : w1b[((size_t)l * DD + k) * HH + (col - HH)];
    w1t[idx] = (__bf16)v;
}

__global__ void prep_w2(const float* __restrict__ w2, __bf16* __restrict__ w2t) {
    int idx = blockIdx.x * 256 + threadIdx.x;      // 8*1024*512 total
    int l   = idx >> 19;
    int rem = idx & 0x7FFFF;
    int col = rem >> 9;
    int k   = rem & 511;
    w2t[idx] = (__bf16)w2[((size_t)l * HH + k) * DD + col];
}

// ---------------------------------------------------------------------------
// Gates: one wave32 per token; 7 dots, shuffle-reduce, sigmoid, tree mixture
// (faithful flat c_nc indexing: mix[leaf] = f(leaf>>2)*f(leaf>>1)*f(leaf)).
// ---------------------------------------------------------------------------
__global__ void gates_kernel(const float* __restrict__ x,
                             const float* __restrict__ nk,
                             const float* __restrict__ nb,
                             float* __restrict__ mix) {
    int lane = threadIdx.x & 31;
    int tok  = blockIdx.x * 8 + (threadIdx.x >> 5);
    const float* xr = x + (size_t)tok * DD;
    float acc[7] = {0.f,0.f,0.f,0.f,0.f,0.f,0.f};
    for (int d = lane; d < DD; d += 32) {
        float xv = xr[d];
        const float* kr = nk + d * 7;
        #pragma unroll
        for (int n = 0; n < 7; ++n) acc[n] += xv * kr[n];
    }
    #pragma unroll
    for (int off = 16; off > 0; off >>= 1) {
        #pragma unroll
        for (int n = 0; n < 7; ++n) acc[n] += __shfl_down(acc[n], off, 32);
    }
    if (lane == 0) {
        float c[7];
        #pragma unroll
        for (int n = 0; n < 7; ++n)
            c[n] = 1.f / (1.f + __expf(-(acc[n] + nb[n])));
        #pragma unroll
        for (int leaf = 0; leaf < NLV; ++leaf) {
            float m = 1.f;
            #pragma unroll
            for (int sh = 2; sh >= 0; --sh) {
                int j = leaf >> sh;
                float cv = c[j >> 1];
                m *= (j & 1) ? (1.f - cv) : cv;
            }
            mix[tok * NLV + leaf] = m;
        }
    }
}

// ---------------------------------------------------------------------------
// Fused FFN: 32 tokens per workgroup (2 WMMA row-tiles), 16 waves.
// Every B-fragment feeds 2 WMMAs (halves L2 weight traffic vs Mtile=16).
//   layer1: wave owns 32 a-cols + 32 b-cols -> 4 N-tiles x 2 M-tiles
//   layer2: wave owns 64 out-cols          -> 4 N-tiles x 2 M-tiles
// ---------------------------------------------------------------------------
__global__ __launch_bounds__(512)
void ffn_kernel(const float* __restrict__ x,
                const __bf16* __restrict__ w1t,
                const __bf16* __restrict__ w2t,
                const float* __restrict__ b1a,
                const float* __restrict__ b1b,
                const float* __restrict__ b2,
                const float* __restrict__ mix,
                float* __restrict__ out) {
    __shared__ __bf16 xs[32 * DD];   // 64 KB bf16 x tile (row-major)
    __shared__ __bf16 hs[32 * HH];   // 32 KB bf16 h tile (row-major)
    __shared__ float  ms[32 * NLV];  // 1 KB mixture tile

    const int tid  = threadIdx.x;
    const int wave = tid >> 5;       // 0..15
    const int lane = tid & 31;
    const int half = lane >> 4;      // K-half held by this lane
    const int ln   = lane & 15;      // M (A) / N (B,C,D) index
    const int tok0 = blockIdx.x * 32;

    for (int i = tid; i < 32 * DD; i += 512) {
        int r = i >> 10, d = i & (DD - 1);
        xs[i] = (__bf16)x[(size_t)(tok0 + r) * DD + d];
    }
    if (tid < 32 * NLV) ms[tid] = mix[tok0 * NLV + tid];
    __syncthreads();

    v8f oacc[2][4];                  // [Mtile][Ntile] persistent across leaves
    #pragma unroll
    for (int m = 0; m < 2; ++m)
        #pragma unroll
        for (int t = 0; t < 4; ++t) oacc[m][t] = (v8f){};

    for (int l = 0; l < NLV; ++l) {
        if (l + 1 < NLV) {           // global_prefetch_b8: warm next leaf
            __builtin_prefetch(w1t + (size_t)(l + 1) * DD * DD, 0, 1);
            __builtin_prefetch(w2t + (size_t)(l + 1) * DD * HH, 0, 1);
        }
        // ----- layer 1: h = (x*W1a + b1a) .* (x*W1b + b1b) -----
        v8f accA[2][2], accB[2][2];  // [Mtile][Ntile]
        #pragma unroll
        for (int m = 0; m < 2; ++m)
            #pragma unroll
            for (int t = 0; t < 2; ++t) { accA[m][t] = (v8f){}; accB[m][t] = (v8f){}; }

        const __bf16* w1l = w1t + (size_t)l * DD * DD;
        for (int kk = 0; kk < DD; kk += 32) {
            v16bf af[2];
            #pragma unroll
            for (int m = 0; m < 2; ++m) {
                int ab = (m * 16 + ln) * DD + kk + half * 16;
                af[m] = cat8(*(const v8bf*)&xs[ab], *(const v8bf*)&xs[ab + 8]);
            }
            #pragma unroll
            for (int t = 0; t < 2; ++t) {
                const __bf16* bpA = w1l + (size_t)(wave * 32 + t * 16 + ln) * DD + kk + half * 16;
                v16bf bA = cat8(*(const v8bf*)bpA, *(const v8bf*)(bpA + 8));
                const __bf16* bpB = w1l + (size_t)(HH + wave * 32 + t * 16 + ln) * DD + kk + half * 16;
                v16bf bB = cat8(*(const v8bf*)bpB, *(const v8bf*)(bpB + 8));
                #pragma unroll
                for (int m = 0; m < 2; ++m) {
                    accA[m][t] = __builtin_amdgcn_wmma_f32_16x16x32_bf16(
                        false, af[m], false, bA, (short)0, accA[m][t], false, false);
                    accB[m][t] = __builtin_amdgcn_wmma_f32_16x16x32_bf16(
                        false, af[m], false, bB, (short)0, accB[m][t], false, false);
                }
            }
        }
        // GLU -> bf16 h tile (C layout: row = m*16 + r + 8*half, col = base + ln)
        #pragma unroll
        for (int t = 0; t < 2; ++t) {
            int col = wave * 32 + t * 16 + ln;
            float ba = b1a[l * HH + col];
            float bb = b1b[l * HH + col];
            #pragma unroll
            for (int m = 0; m < 2; ++m)
                #pragma unroll
                for (int r = 0; r < 8; ++r) {
                    float v = (accA[m][t][r] + ba) * (accB[m][t][r] + bb);
                    hs[(m * 16 + r + 8 * half) * HH + col] = (__bf16)v;
                }
        }
        __syncthreads();

        // ----- layer 2: y = h * W2[l]; out += mix[:,l] * (y + b2[l]) -----
        v8f y[2][4];
        #pragma unroll
        for (int m = 0; m < 2; ++m)
            #pragma unroll
            for (int t = 0; t < 4; ++t) y[m][t] = (v8f){};

        const __bf16* w2l = w2t + (size_t)l * DD * HH;
        for (int kk = 0; kk < HH; kk += 32) {
            v16bf af[2];
            #pragma unroll
            for (int m = 0; m < 2; ++m) {
                int ab = (m * 16 + ln) * HH + kk + half * 16;
                af[m] = cat8(*(const v8bf*)&hs[ab], *(const v8bf*)&hs[ab + 8]);
            }
            #pragma unroll
            for (int t = 0; t < 4; ++t) {
                const __bf16* bp = w2l + (size_t)(wave * 64 + t * 16 + ln) * HH + kk + half * 16;
                v16bf bf = cat8(*(const v8bf*)bp, *(const v8bf*)(bp + 8));
                #pragma unroll
                for (int m = 0; m < 2; ++m)
                    y[m][t] = __builtin_amdgcn_wmma_f32_16x16x32_bf16(
                        false, af[m], false, bf, (short)0, y[m][t], false, false);
            }
        }
        #pragma unroll
        for (int t = 0; t < 4; ++t) {
            float bv = b2[l * DD + wave * 64 + t * 16 + ln];
            #pragma unroll
            for (int m = 0; m < 2; ++m)
                #pragma unroll
                for (int r = 0; r < 8; ++r) {
                    float s = ms[(m * 16 + r + 8 * half) * NLV + l];
                    oacc[m][t][r] += s * (y[m][t][r] + bv);
                }
        }
        __syncthreads();   // hs reused next leaf
    }

    #pragma unroll
    for (int t = 0; t < 4; ++t) {
        int col = wave * 64 + t * 16 + ln;
        #pragma unroll
        for (int m = 0; m < 2; ++m)
            #pragma unroll
            for (int r = 0; r < 8; ++r)
                out[(size_t)(tok0 + m * 16 + r + 8 * half) * DD + col] = oacc[m][t][r];
    }
}

// ---------------------------------------------------------------------------
extern "C" void kernel_launch(void* const* d_in, const int* in_sizes, int n_in,
                              void* d_out, int out_size, void* d_ws, size_t ws_size,
                              hipStream_t stream) {
    const float* x   = (const float*)d_in[0];
    const float* nk  = (const float*)d_in[1];
    const float* nb  = (const float*)d_in[2];
    const float* w1a = (const float*)d_in[3];
    const float* w1b = (const float*)d_in[4];
    const float* b1a = (const float*)d_in[5];
    const float* b1b = (const float*)d_in[6];
    const float* w2  = (const float*)d_in[7];
    const float* b2  = (const float*)d_in[8];
    float* out = (float*)d_out;

    char* ws = (char*)d_ws;
    float*  mixb = (float*)ws;                                        // 256 KB
    __bf16* w1t  = (__bf16*)(ws + (size_t)NTOK * NLV * 4);            // 16 MB
    __bf16* w2t  = (__bf16*)((char*)w1t + (size_t)NLV * DD * DD * 2); // 8 MB

    prep_w1<<<(NLV * DD * DD) / 256, 256, 0, stream>>>(w1a, w1b, w1t);
    prep_w2<<<(NLV * DD * HH) / 256, 256, 0, stream>>>(w2, w2t);
    gates_kernel<<<NTOK / 8, 256, 0, stream>>>(x, nk, nb, mixb);
    ffn_kernel<<<NTOK / 32, 512, 0, stream>>>(x, w1t, w2t, b1a, b1b, b2, mixb, out);
}
